// Vim_48816598286475
// MI455X (gfx1250) — matrix operand order repaired and checked by reading, not compiled, not run on hardware
//
#include <hip/hip_runtime.h>
#include <hip/hip_bf16.h>
#include <cstdint>
#include <cstddef>

// ---------------- model constants ----------------
#define DIMC      256
#define DT_RANK   32
#define D_STATE   16
#define DEPTH     12
#define IMGSZ     224
#define PATCHSZ   16
#define GRIDSZ    14
#define NPATCH    196          // 14*14
#define PATCH_DIM 768          // 3*16*16
#define BATCH     8
#define SEQ       197          // NPATCH + 1 (cls)
#define ROWS      (BATCH*SEQ)  // 1576
#define ROWS_PAD  1600         // multiple of 32
#define KP_TP     224          // 196 padded to mult of 32
#define OUTROWS   (BATCH*DIMC) // 2048
#define NPIX      (IMGSZ*IMGSZ) // 50176

typedef __bf16 bf16_t;
typedef bf16_t v16bf __attribute__((ext_vector_type(16)));
typedef bf16_t v8bf  __attribute__((ext_vector_type(8)));
typedef float  v8f   __attribute__((ext_vector_type(8)));

// epilogue op flags
#define OP_SOFTPLUS 1
#define OP_NTSTORE  2   // non-temporal output stores (streaming, write-once)

// ---------------- device helpers ----------------
__device__ __forceinline__ float softplusf(float v) {
  return v > 20.f ? v : log1pf(__expf(v));
}

// A-fragment (16x32 bf16, row-major source, this lane covers row m=lane&15):
// VGPR0..7 hold K = g*16..g*16+15 (g=lane>>4)
__device__ __forceinline__ v16bf load_afrag(const bf16_t* p) {
  v8bf lo = *(const v8bf*)(p);        // k = g*16 .. g*16+7
  v8bf hi = *(const v8bf*)(p + 8);    // k = g*16+8 .. g*16+15
  v16bf r;
#pragma unroll
  for (int i = 0; i < 8; ++i) { r[i] = lo[i]; r[i + 8] = hi[i]; }
  return r;
}

__device__ __forceinline__ void store_c(float* C, bf16_t* Cbf, size_t ldc,
                                        size_t row, int col, float v, int op) {
  if (op & OP_SOFTPLUS) v = softplusf(v);
  if (op & OP_NTSTORE)  __builtin_nontemporal_store(v, &C[row * ldc + col]);
  else                  C[row * ldc + col] = v;
  if (Cbf) Cbf[row * ldc + col] = (bf16_t)v;
}

// ---------------- generic bf16 WMMA GEMM ----------------
// C(M,N) f32 = A(M,Kp) bf16 row-major  x  Bt(N,Kp) bf16 row-major (B transposed)
//            + bias(N) [optional], epilogue op flags,
//            optional secondary bf16 copy of C.
// One wave computes a 32x32 tile (2x2 fragments of 16x16). M,N mult of 32,
// Kp mult of 32. Wave-uniform guard only (EXEC stays all-1s for WMMA).
__global__ __launch_bounds__(256) void gemm_bf16_wmma(
    const bf16_t* __restrict__ A, int lda,
    const bf16_t* __restrict__ Bt, int ldb,
    const float* __restrict__ bias,
    float* __restrict__ C, bf16_t* __restrict__ Cbf, size_t ldc,
    int MT2, int NT2, int kTiles, int op)
{
  int wave = threadIdx.x >> 5;
  int t = blockIdx.x * 8 + wave;
  if (t >= MT2 * NT2) return;              // wave-uniform
  int mt = t / NT2, nt = t % NT2;
  int lane = threadIdx.x & 31;
  int li = lane & 15, g = lane >> 4;

  const bf16_t* a0 = A + (size_t)(mt * 32 + li) * lda + g * 16;
  const bf16_t* a1 = a0 + (size_t)16 * lda;
  // B fragment: lane covers column n=li; lanes 0-15 hold K=0..15, 16-31 K=16..31
  const bf16_t* b0 = Bt + (size_t)(nt * 32 + li) * ldb + g * 16;
  const bf16_t* b1 = b0 + (size_t)16 * ldb;

  v8f acc00 = {}, acc01 = {}, acc10 = {}, acc11 = {};
  for (int kt = 0; kt < kTiles; ++kt) {
    int ko = kt * 32;
    __builtin_prefetch(b0 + ko + 64, 0, 1);   // global_prefetch_b8 (gfx1250)
    __builtin_prefetch(a0 + ko + 64, 0, 1);
    v16bf fa0 = load_afrag(a0 + ko);
    v16bf fa1 = load_afrag(a1 + ko);
    v16bf fb0 = *(const v16bf*)(b0 + ko);
    v16bf fb1 = *(const v16bf*)(b1 + ko);
    acc00 = __builtin_amdgcn_wmma_f32_16x16x32_bf16(false, fa0, false, fb0,
                                                    (short)0, acc00, false, false);
    acc01 = __builtin_amdgcn_wmma_f32_16x16x32_bf16(false, fa0, false, fb1,
                                                    (short)0, acc01, false, false);
    acc10 = __builtin_amdgcn_wmma_f32_16x16x32_bf16(false, fa1, false, fb0,
                                                    (short)0, acc10, false, false);
    acc11 = __builtin_amdgcn_wmma_f32_16x16x32_bf16(false, fa1, false, fb1,
                                                    (short)0, acc11, false, false);
  }
  // C/D layout: VGPR r -> M = r (lanes 0-15), M = 8+r (lanes 16-31); N = lane&15
  int n0 = nt * 32 + li;
  int n1 = n0 + 16;
  float bi0 = bias ? bias[n0] : 0.f;
  float bi1 = bias ? bias[n1] : 0.f;
  size_t rbase = (size_t)mt * 32 + g * 8;
#pragma unroll
  for (int r = 0; r < 8; ++r) {
    store_c(C, Cbf, ldc, rbase + r,      n0, acc00[r] + bi0, op);
    store_c(C, Cbf, ldc, rbase + r,      n1, acc01[r] + bi1, op);
    store_c(C, Cbf, ldc, rbase + 16 + r, n0, acc10[r] + bi0, op);
    store_c(C, Cbf, ldc, rbase + 16 + r, n1, acc11[r] + bi1, op);
  }
}

// ---------------- weight convert (+optional transpose, K padded) ----------------
// dst (N, Kp) bf16 : transpose ? src(K,N) : src(N,K); k>=K zero-filled.
__global__ void convert_w_kernel(const float* __restrict__ src, bf16_t* __restrict__ dst,
                                 int K, int N, int Kp, int transpose) {
  long idx = (long)blockIdx.x * 256 + threadIdx.x;
  long total = (long)N * Kp;
  if (idx >= total) return;
  int n = (int)(idx / Kp), k = (int)(idx % Kp);
  float v = 0.f;
  if (k < K) v = transpose ? src[(size_t)k * N + n] : src[(size_t)n * K + k];
  dst[idx] = (bf16_t)v;
}

// ---------------- patchify: x(8,3,224,224) -> patches bf16 (1568, 768) ----------------
__global__ void patchify_kernel(const float* __restrict__ x, bf16_t* __restrict__ out) {
  long idx = (long)blockIdx.x * 256 + threadIdx.x;
  if (idx >= (long)BATCH * NPATCH * PATCH_DIM) return;
  int col = (int)(idx % PATCH_DIM);
  long row = idx / PATCH_DIM;
  int b = (int)(row / NPATCH);
  int p = (int)(row % NPATCH);
  int h = p / GRIDSZ, w = p % GRIDSZ;
  int p1 = col / (PATCHSZ * 3);
  int rem = col % (PATCHSZ * 3);
  int p2 = rem / 3, c = rem % 3;
  size_t src = (((size_t)b * 3 + c) * IMGSZ + (h * PATCHSZ + p1)) * IMGSZ
             + (w * PATCHSZ + p2);
  out[idx] = (bf16_t)x[src];
}

// ---------------- assemble h: cls + tok -> h f32 (ROWS,256) ----------------
__global__ void assemble_h_kernel(const float* __restrict__ tok,
                                  const float* __restrict__ cls,
                                  float* __restrict__ h) {
  long idx = (long)blockIdx.x * 256 + threadIdx.x;
  if (idx >= (long)ROWS * DIMC) return;
  int d = (int)(idx % DIMC);
  long r = idx / DIMC;
  int b = (int)(r / SEQ);
  int i = (int)(r % SEQ);
  h[idx] = (i == 0) ? cls[d] : tok[((size_t)b * NPATCH + (i - 1)) * DIMC + d];
}

// ---------------- layernorm: h f32 -> hn bf16, one wave per row ----------------
__global__ __launch_bounds__(256) void ln_kernel(const float* __restrict__ h,
                                                 const float* __restrict__ w,
                                                 const float* __restrict__ b,
                                                 bf16_t* __restrict__ out) {
  int row = blockIdx.x * 8 + (threadIdx.x >> 5);
  if (row >= ROWS) return;
  int lane = threadIdx.x & 31;
  const float* p = h + (size_t)row * DIMC + lane * 8;
  float v[8];
#pragma unroll
  for (int i = 0; i < 8; ++i) v[i] = p[i];
  float s = 0.f;
#pragma unroll
  for (int i = 0; i < 8; ++i) s += v[i];
#pragma unroll
  for (int m = 16; m > 0; m >>= 1) s += __shfl_xor(s, m, 32);
  float mu = s * (1.f / DIMC);
  float ss = 0.f;
#pragma unroll
  for (int i = 0; i < 8; ++i) { float d0 = v[i] - mu; ss += d0 * d0; }
#pragma unroll
  for (int m = 16; m > 0; m >>= 1) ss += __shfl_xor(ss, m, 32);
  float rinv = rsqrtf(ss * (1.f / DIMC) + 1e-5f);
  bf16_t* o = out + (size_t)row * DIMC + lane * 8;
#pragma unroll
  for (int i = 0; i < 8; ++i)
    o[i] = (bf16_t)((v[i] - mu) * rinv * w[lane * 8 + i] + b[lane * 8 + i]);
}

// ---------------- selective scan: one thread per (batch, channel) ----------------
// x_t[n] = exp(delta*A[n]) * x_{t-1}[n] + delta*u*B_t[n];  y = <x_t, C_t> + u*D
__global__ __launch_bounds__(256) void ssm_kernel(const float* __restrict__ delta,
                                                  const float* __restrict__ u,
                                                  const float* __restrict__ dbc,
                                                  const float* __restrict__ A_log,
                                                  const float* __restrict__ Dp,
                                                  float* __restrict__ y2) {
  int b = blockIdx.x;
  int d = threadIdx.x;
  float A[D_STATE];
#pragma unroll
  for (int n = 0; n < D_STATE; ++n) A[n] = -__expf(A_log[(size_t)d * D_STATE + n]);
  float xs[D_STATE];
#pragma unroll
  for (int n = 0; n < D_STATE; ++n) xs[n] = 0.f;
  float Dd = Dp[d];
  for (int l = 0; l < SEQ; ++l) {
    size_t r = (size_t)b * SEQ + l;
    float dlt = delta[r * DIMC + d];
    float uu  = u[r * DIMC + d];
    float du  = dlt * uu;
    const float* Brow = dbc + r * 64 + DT_RANK;           // cols 32..47
    const float* Crow = dbc + r * 64 + DT_RANK + D_STATE; // cols 48..63
    float acc = 0.f;
#pragma unroll
    for (int n = 0; n < D_STATE; ++n) {
      float sfac = __expf(dlt * A[n]);
      xs[n] = xs[n] * sfac + du * Brow[n];
      acc += xs[n] * Crow[n];
    }
    y2[r * DIMC + d] = acc + uu * Dd;
  }
}

// ---------------- combine: h += (z1 + y2) * silu(z1) ----------------
__global__ void combine_kernel(float* __restrict__ h,
                               const float* __restrict__ z1,
                               const float* __restrict__ y2) {
  long idx = (long)blockIdx.x * 256 + threadIdx.x;
  if (idx >= (long)ROWS * DIMC) return;
  float z = z1[idx];
  float sl = z / (1.f + __expf(-z));       // silu(z1)
  h[idx] += (z + y2[idx]) * sl;
}

// ---------------- build Y bf16 (2048, 224): Y[b*256+d][l] = h[b,1+l,d] ----------------
__global__ void build_y_kernel(const float* __restrict__ h, bf16_t* __restrict__ Y) {
  long idx = (long)blockIdx.x * 256 + threadIdx.x;
  if (idx >= (long)OUTROWS * KP_TP) return;
  int l = (int)(idx % KP_TP);
  long row = idx / KP_TP;
  int b = (int)(row / DIMC);
  int d = (int)(row % DIMC);
  float v = 0.f;
  if (l < NPATCH) v = h[((size_t)b * SEQ + 1 + l) * DIMC + d];
  Y[idx] = (bf16_t)v;
}

// ---------------- host side ----------------
static inline void launch_gemm(const bf16_t* A, int lda, const bf16_t* Bt, int ldb,
                               const float* bias, float* C, bf16_t* Cbf, size_t ldc,
                               int M, int N, int K, int op, hipStream_t s) {
  int MT2 = M / 32, NT2 = N / 32, kT = K / 32;
  int tiles = MT2 * NT2;
  int blocks = (tiles + 7) / 8;
  gemm_bf16_wmma<<<blocks, 256, 0, s>>>(A, lda, Bt, ldb, bias, C, Cbf, ldc,
                                        MT2, NT2, kT, op);
}

static inline int grid1d(long total) { return (int)((total + 255) / 256); }

extern "C" void kernel_launch(void* const* d_in, const int* in_sizes, int n_in,
                              void* d_out, int out_size, void* d_ws, size_t ws_size,
                              hipStream_t stream) {
  const float* x         = (const float*)d_in[0];
  const float* patch_w   = (const float*)d_in[1];
  const float* patch_b   = (const float*)d_in[2];
  const float* cls_token = (const float*)d_in[3];
  const float* norm_w    = (const float*)d_in[4];
  const float* norm_b    = (const float*)d_in[5];
  const float* proj_w    = (const float*)d_in[6];
  const float* proj_b    = (const float*)d_in[7];
  const float* bconv_w   = (const float*)d_in[8];
  const float* bconv_b   = (const float*)d_in[9];
  const float* dbc_w     = (const float*)d_in[10];
  const float* dt_w      = (const float*)d_in[11];
  const float* dt_b      = (const float*)d_in[12];
  const float* A_log     = (const float*)d_in[13];
  const float* Dp        = (const float*)d_in[14];
  const float* tp_w      = (const float*)d_in[15];
  const float* tp_b      = (const float*)d_in[16];
  float* out = (float*)d_out;

  // ---- workspace carve (256B aligned) ----
  uint8_t* wp = (uint8_t*)d_ws;
  auto alloc = [&](size_t bytes) -> void* {
    void* r = (void*)wp;
    wp += (bytes + 255) & ~(size_t)255;
    return r;
  };
  bf16_t* patches  = (bf16_t*)alloc((size_t)BATCH * NPATCH * PATCH_DIM * 2);
  bf16_t* patch_wt = (bf16_t*)alloc((size_t)DIMC * PATCH_DIM * 2);
  float*  tok      = (float*)alloc((size_t)BATCH * NPATCH * DIMC * 4);
  float*  h        = (float*)alloc((size_t)ROWS_PAD * DIMC * 4);
  bf16_t* hn       = (bf16_t*)alloc((size_t)ROWS_PAD * DIMC * 2);
  float*  z1       = (float*)alloc((size_t)ROWS_PAD * DIMC * 4);
  bf16_t* z1b      = (bf16_t*)alloc((size_t)ROWS_PAD * DIMC * 2);
  float*  x2       = (float*)alloc((size_t)ROWS_PAD * DIMC * 4);
  bf16_t* x2b      = (bf16_t*)alloc((size_t)ROWS_PAD * DIMC * 2);
  float*  dbc      = (float*)alloc((size_t)ROWS_PAD * 64 * 4);
  bf16_t* dbcb     = (bf16_t*)alloc((size_t)ROWS_PAD * 64 * 2);
  float*  delta    = (float*)alloc((size_t)ROWS_PAD * DIMC * 4);
  float*  y2       = (float*)alloc((size_t)ROWS_PAD * DIMC * 4);
  bf16_t* projwt   = (bf16_t*)alloc((size_t)DEPTH * DIMC * DIMC * 2);
  bf16_t* bconvb   = (bf16_t*)alloc((size_t)DEPTH * DIMC * DIMC * 2);
  bf16_t* dbcwt    = (bf16_t*)alloc((size_t)DEPTH * 64 * DIMC * 2);
  bf16_t* dtwt     = (bf16_t*)alloc((size_t)DEPTH * DIMC * DT_RANK * 2);
  bf16_t* Ybuf     = (bf16_t*)alloc((size_t)OUTROWS * KP_TP * 2);
  bf16_t* tpwt     = (bf16_t*)alloc((size_t)NPIX * KP_TP * 2);

  // ---- weight conversions (f32 -> bf16, B transposed to (N,Kp)) ----
  convert_w_kernel<<<grid1d((long)DIMC * PATCH_DIM), 256, 0, stream>>>(
      patch_w, patch_wt, PATCH_DIM, DIMC, PATCH_DIM, 1);
  for (int l = 0; l < DEPTH; ++l) {
    convert_w_kernel<<<grid1d((long)DIMC * DIMC), 256, 0, stream>>>(
        proj_w + (size_t)l * DIMC * DIMC, projwt + (size_t)l * DIMC * DIMC,
        DIMC, DIMC, DIMC, 1);
    // bconv: x2 = x1 @ bw^T  =>  Bt == bw (out,in) as-is
    convert_w_kernel<<<grid1d((long)DIMC * DIMC), 256, 0, stream>>>(
        bconv_w + (size_t)l * DIMC * DIMC, bconvb + (size_t)l * DIMC * DIMC,
        DIMC, DIMC, DIMC, 0);
    convert_w_kernel<<<grid1d((long)64 * DIMC), 256, 0, stream>>>(
        dbc_w + (size_t)l * DIMC * 64, dbcwt + (size_t)l * 64 * DIMC,
        DIMC, 64, DIMC, 1);
    convert_w_kernel<<<grid1d((long)DIMC * DT_RANK), 256, 0, stream>>>(
        dt_w + (size_t)l * DT_RANK * DIMC, dtwt + (size_t)l * DIMC * DT_RANK,
        DT_RANK, DIMC, DT_RANK, 1);
  }
  convert_w_kernel<<<grid1d((long)NPIX * KP_TP), 256, 0, stream>>>(
      tp_w, tpwt, NPATCH, NPIX, KP_TP, 1);

  // ---- patch embedding ----
  patchify_kernel<<<grid1d((long)BATCH * NPATCH * PATCH_DIM), 256, 0, stream>>>(x, patches);
  launch_gemm(patches, PATCH_DIM, patch_wt, PATCH_DIM, patch_b,
              tok, nullptr, DIMC, BATCH * NPATCH, DIMC, PATCH_DIM, 0, stream);
  assemble_h_kernel<<<grid1d((long)ROWS * DIMC), 256, 0, stream>>>(tok, cls_token, h);

  // ---- 12 layers ----
  for (int l = 0; l < DEPTH; ++l) {
    ln_kernel<<<(ROWS + 7) / 8, 256, 0, stream>>>(
        h, norm_w + (size_t)l * DIMC, norm_b + (size_t)l * DIMC, hn);
    // z1 = hn @ proj_w + proj_b
    launch_gemm(hn, DIMC, projwt + (size_t)l * DIMC * DIMC, DIMC,
                proj_b + (size_t)l * DIMC, z1, z1b, DIMC,
                ROWS_PAD, DIMC, DIMC, 0, stream);
    // x2 = z1 @ bconv_w^T + bconv_b
    launch_gemm(z1b, DIMC, bconvb + (size_t)l * DIMC * DIMC, DIMC,
                bconv_b + (size_t)l * DIMC, x2, x2b, DIMC,
                ROWS_PAD, DIMC, DIMC, 0, stream);
    // dbc = x2 @ deltaBC_w   -> [delta32 | B16 | C16]
    launch_gemm(x2b, DIMC, dbcwt + (size_t)l * 64 * DIMC, DIMC,
                nullptr, dbc, dbcb, 64,
                ROWS_PAD, 64, DIMC, 0, stream);
    // delta = softplus(dbc[:, :32] @ dt_w + dt_b)  (K=32 reads only first 32 cols)
    launch_gemm(dbcb, 64, dtwt + (size_t)l * DIMC * DT_RANK, DT_RANK,
                dt_b + (size_t)l * DIMC, delta, nullptr, DIMC,
                ROWS_PAD, DIMC, DT_RANK, OP_SOFTPLUS, stream);
    // selective scan
    ssm_kernel<<<BATCH, DIMC, 0, stream>>>(
        delta, x2, dbc, A_log + (size_t)l * DIMC * D_STATE,
        Dp + (size_t)l * DIMC, y2);
    // h += (z1 + y2) * silu(z1)
    combine_kernel<<<grid1d((long)ROWS * DIMC), 256, 0, stream>>>(h, z1, y2);
  }

  // ---- head: out(2048, 50176) = Y(2048,224) @ tpwt^T + tp_b ----
  // 411 MB write-once output: non-temporal stores keep the 22.5 MB bf16
  // weight + Y resident in the 192 MB L2 across all M-tiles.
  build_y_kernel<<<grid1d((long)OUTROWS * KP_TP), 256, 0, stream>>>(h, Ybuf);
  launch_gemm(Ybuf, KP_TP, tpwt, KP_TP, tp_b,
              out, nullptr, (size_t)NPIX, OUTROWS, NPIX, KP_TP, OP_NTSTORE, stream);

  (void)in_sizes; (void)n_in; (void)out_size; (void)ws_size;
}
